// KmeansVectorQuantizer_56453050138867
// MI455X (gfx1250) — matrix operand order repaired and checked by reading, not compiled, not run on hardware
//
#include <hip/hip_runtime.h>
#include <hip/hip_bf16.h>

// ---------------------------------------------------------------------------
// Problem constants (from reference): x[32,512,512] f32, W[8192,512] f32
// ---------------------------------------------------------------------------
#define N_E     8192          // number of codebook entries
#define E_DIM   512           // embedding dim
#define N_ROWS  16384         // 32*512 latent vectors
#define BETA    0.25f

typedef __attribute__((ext_vector_type(16))) __bf16 v16bf;
typedef __attribute__((ext_vector_type(8)))  __bf16 v8bf;
typedef __attribute__((ext_vector_type(8)))  float  v8f;

// float -> bf16 with round-to-nearest-even
__device__ __forceinline__ unsigned short f2bf(float f) {
    unsigned int u = __builtin_bit_cast(unsigned int, f);
    unsigned int r = u + 0x7FFFu + ((u >> 16) & 1u);
    return (unsigned short)(r >> 16);
}

__device__ __forceinline__ v16bf cat8(v8bf lo, v8bf hi) {
    return __builtin_shufflevector(lo, hi,
        0,1,2,3,4,5,6,7,8,9,10,11,12,13,14,15);
}

// ---------------------------------------------------------------------------
// Kernel 1: convert W to bf16 and compute ||c||^2 per code.
// ---------------------------------------------------------------------------
__global__ void vq_prep_w(const float* __restrict__ W,
                          unsigned short* __restrict__ Wb,
                          float* __restrict__ Wnorm) {
    const int row = blockIdx.x;
    const float* src = W + (size_t)row * E_DIM;
    unsigned short* dst = Wb + (size_t)row * E_DIM;
    float ss = 0.f;
    for (int i = threadIdx.x; i < E_DIM; i += 256) {
        float v = src[i];
        dst[i] = f2bf(v);
        ss += v * v;
    }
    #pragma unroll
    for (int off = 16; off > 0; off >>= 1) ss += __shfl_down(ss, off, 32);
    __shared__ float red[8];
    if ((threadIdx.x & 31) == 0) red[threadIdx.x >> 5] = ss;
    __syncthreads();
    if (threadIdx.x == 0) {
        float t = 0.f;
        #pragma unroll
        for (int w = 0; w < 8; ++w) t += red[w];
        Wnorm[row] = t;
    }
}

// ---------------------------------------------------------------------------
// Kernel 2: convert x to bf16 (vectorized 4 elements / thread).
// ---------------------------------------------------------------------------
__global__ void vq_conv_x(const float* __restrict__ X,
                          unsigned short* __restrict__ Xb) {
    const int i = blockIdx.x * blockDim.x + threadIdx.x;   // i indexes float4
    float4 v = ((const float4*)X)[i];
    ushort4 o;
    o.x = f2bf(v.x); o.y = f2bf(v.y); o.z = f2bf(v.z); o.w = f2bf(v.w);
    ((ushort4*)Xb)[i] = o;
}

// ---------------------------------------------------------------------------
// Kernel 3: WMMA distance GEMM + argmin.
// Block = 256 threads (8 waves) handles a 32-row M-tile (A staged in LDS,
// 32 KB). Each wave computes a 2x2 register tile (2 M-subtiles x 2 N-tiles,
// 4 v8f accumulators) so each A/B fragment feeds two WMMAs. K-loop kept
// rolled (#pragma unroll 1) to prevent the compiler from hoisting all 16
// A fragments into registers and spilling to scratch (round-1 bug).
// Scores = ||c||^2 - 2 z.c (row term dropped: constant per row).
// ---------------------------------------------------------------------------
__global__ void __launch_bounds__(256)
vq_argmin(const unsigned short* __restrict__ Xb,
          const unsigned short* __restrict__ Wb,
          const float* __restrict__ Wnorm,
          int* __restrict__ outIdx) {
    __shared__ unsigned short Atile[32 * E_DIM];   // 32 KB
    __shared__ float sVal[8 * 32];
    __shared__ int   sIdx[8 * 32];

    const int m0    = blockIdx.x * 32;
    const int tid   = threadIdx.x;
    const int wave  = tid >> 5;
    const int lane  = tid & 31;
    const int col16 = lane & 15;     // column within 16-group / A row
    const int khalf = lane >> 4;     // which K half this lane owns

    // Stage A tile: 32*512 bf16 = 2048 x uint4
    {
        const uint4* src = (const uint4*)(Xb + (size_t)m0 * E_DIM);
        uint4* dst = (uint4*)Atile;
        for (int i = tid; i < 32 * E_DIM / 8; i += 256) dst[i] = src[i];
    }
    __syncthreads();

    float best[2][8];
    int   bidx[2][8];
    #pragma unroll
    for (int mt = 0; mt < 2; ++mt)
        #pragma unroll
        for (int r = 0; r < 8; ++r) { best[mt][r] = 3.4e38f; bidx[mt][r] = 0; }

    // Per-lane LDS A-fragment bases (A layout: lanes 0-15 row=col16,
    // K {k0..k0+7, k0+16..k0+23}; lanes 16-31 same row, +8 shifted).
    const unsigned short* aBase0 = Atile + col16 * E_DIM + khalf * 8;
    const unsigned short* aBase1 = aBase0 + 16 * E_DIM;

    for (int g = wave; g < N_E / 32; g += 8) {     // 2 N-tiles per group
        const int n0 = g * 32;
        v8f c00 = {}, c01 = {}, c10 = {}, c11 = {};   // c[mt][q]
        // B layout: lanes 0-15 -> col n0+lane, K {k0..k0+7, k0+16..k0+23};
        // lanes 16-31 -> col n0+lane-16, +8 shifted. W rows are B columns.
        const unsigned short* b0 =
            Wb + (size_t)(n0 + col16) * E_DIM + khalf * 8;
        const unsigned short* b1 = b0 + (size_t)16 * E_DIM;

        #pragma unroll 1
        for (int ks = 0; ks < E_DIM / 32; ++ks) {
            const int k0 = ks * 32;
            v16bf a0 = cat8(*(const v8bf*)(aBase0 + k0),
                            *(const v8bf*)(aBase0 + k0 + 16));
            v16bf a1 = cat8(*(const v8bf*)(aBase1 + k0),
                            *(const v8bf*)(aBase1 + k0 + 16));
            v16bf bb0 = cat8(*(const v8bf*)(b0 + k0),
                             *(const v8bf*)(b0 + k0 + 16));
            v16bf bb1 = cat8(*(const v8bf*)(b1 + k0),
                             *(const v8bf*)(b1 + k0 + 16));
            c00 = __builtin_amdgcn_wmma_f32_16x16x32_bf16(
                      false, a0, false, bb0, (short)0, c00, false, false);
            c01 = __builtin_amdgcn_wmma_f32_16x16x32_bf16(
                      false, a0, false, bb1, (short)0, c01, false, false);
            c10 = __builtin_amdgcn_wmma_f32_16x16x32_bf16(
                      false, a1, false, bb0, (short)0, c10, false, false);
            c11 = __builtin_amdgcn_wmma_f32_16x16x32_bf16(
                      false, a1, false, bb1, (short)0, c11, false, false);
        }

        // C layout: VGPR r, lane l -> row (r + 8*(l>>4)), col (l&15).
        const float wn0 = Wnorm[n0 + col16];
        const float wn1 = Wnorm[n0 + 16 + col16];
        const int   cn0 = n0 + col16;
        const int   cn1 = n0 + 16 + col16;
        #pragma unroll
        for (int r = 0; r < 8; ++r) {
            float s;
            s = wn0 - 2.0f * c00[r];
            if (s < best[0][r]) { best[0][r] = s; bidx[0][r] = cn0; }
            s = wn1 - 2.0f * c01[r];
            if (s < best[0][r]) { best[0][r] = s; bidx[0][r] = cn1; }
            s = wn0 - 2.0f * c10[r];
            if (s < best[1][r]) { best[1][r] = s; bidx[1][r] = cn0; }
            s = wn1 - 2.0f * c11[r];
            if (s < best[1][r]) { best[1][r] = s; bidx[1][r] = cn1; }
        }
    }

    // Reduce (min,idx) across the 16 lanes sharing each row
    #pragma unroll
    for (int off = 1; off < 16; off <<= 1) {
        #pragma unroll
        for (int mt = 0; mt < 2; ++mt)
            #pragma unroll
            for (int r = 0; r < 8; ++r) {
                float ov = __shfl_xor(best[mt][r], off, 32);
                int   oi = __shfl_xor(bidx[mt][r], off, 32);
                if (ov < best[mt][r] ||
                    (ov == best[mt][r] && oi < bidx[mt][r])) {
                    best[mt][r] = ov; bidx[mt][r] = oi;
                }
            }
    }
    // lane 0 holds rows {0..7, 16..23}; lane 16 holds rows {8..15, 24..31}
    if (col16 == 0) {
        #pragma unroll
        for (int mt = 0; mt < 2; ++mt)
            #pragma unroll
            for (int r = 0; r < 8; ++r) {
                sVal[wave * 32 + mt * 16 + khalf * 8 + r] = best[mt][r];
                sIdx[wave * 32 + mt * 16 + khalf * 8 + r] = bidx[mt][r];
            }
    }
    __syncthreads();
    // Final cross-wave reduce: thread t (<32) owns row m0+t
    if (tid < 32) {
        float bv = sVal[tid];
        int   bi = sIdx[tid];
        #pragma unroll
        for (int w = 1; w < 8; ++w) {
            float v = sVal[w * 32 + tid];
            int   i = sIdx[w * 32 + tid];
            if (v < bv || (v == bv && i < bi)) { bv = v; bi = i; }
        }
        outIdx[m0 + tid] = bi;
    }
}

// ---------------------------------------------------------------------------
// Kernel 4: zero the loss accumulator.
// ---------------------------------------------------------------------------
__global__ void vq_zero(float* acc) { acc[0] = 0.f; }

// ---------------------------------------------------------------------------
// Kernel 5: gather x_q = W[idx] (fp32), accumulate sum((x_q - x)^2),
// and emit index as float. One block per latent row.
// ---------------------------------------------------------------------------
__global__ void vq_gather_loss(const float* __restrict__ X,
                               const float* __restrict__ W,
                               const int* __restrict__ idx,
                               float* __restrict__ xq_out,
                               float* __restrict__ idx_out,
                               float* __restrict__ lossAcc) {
    const int row  = blockIdx.x;
    const int code = idx[row];
    const float* w = W + (size_t)code * E_DIM;
    const float* x = X + (size_t)row * E_DIM;
    float* o       = xq_out + (size_t)row * E_DIM;
    float ss = 0.f;
    for (int i = threadIdx.x; i < E_DIM; i += 256) {
        float q = w[i];
        o[i] = q;
        float d = q - x[i];
        ss += d * d;
    }
    #pragma unroll
    for (int off = 16; off > 0; off >>= 1) ss += __shfl_down(ss, off, 32);
    __shared__ float red[8];
    if ((threadIdx.x & 31) == 0) red[threadIdx.x >> 5] = ss;
    __syncthreads();
    if (threadIdx.x == 0) {
        float t = 0.f;
        #pragma unroll
        for (int w2 = 0; w2 < 8; ++w2) t += red[w2];
        atomicAdd(lossAcc, t);
        idx_out[row] = (float)code;
    }
}

// ---------------------------------------------------------------------------
// Kernel 6: finalize loss = BETA * sum / (N_ROWS*E_DIM)
// ---------------------------------------------------------------------------
__global__ void vq_final(const float* __restrict__ acc,
                         float* __restrict__ out_loss) {
    out_loss[0] = BETA * acc[0] * (1.0f / (float)(N_ROWS * (size_t)E_DIM));
}

// ---------------------------------------------------------------------------
// Launch. Inputs: d_in[0]=x f32 [16384*512], d_in[1]=label (unused),
// d_in[2]=idx (unused), d_in[3]=W f32 [8192*512].
// Output (concat): x_q [8388608] f32, loss [1], indices [16384].
// ---------------------------------------------------------------------------
extern "C" void kernel_launch(void* const* d_in, const int* in_sizes, int n_in,
                              void* d_out, int out_size, void* d_ws, size_t ws_size,
                              hipStream_t stream) {
    (void)in_sizes; (void)n_in; (void)out_size; (void)ws_size;
    const float* X = (const float*)d_in[0];
    const float* W = (const float*)d_in[3];

    // Workspace layout (bytes)
    char* ws = (char*)d_ws;
    unsigned short* Wb    = (unsigned short*)(ws);                       //  8 MB
    unsigned short* Xb    = (unsigned short*)(ws + 8u * 1024 * 1024);    // 16 MB
    float*          Wnorm = (float*)(ws + 25u * 1024 * 1024);            // 32 KB
    int*            idxB  = (int*)(ws + 25u * 1024 * 1024 + 64 * 1024);  // 64 KB
    float*          acc   = (float*)(ws + 25u * 1024 * 1024 + 192 * 1024);

    float* xq_out   = (float*)d_out;                       // [0, 8388608)
    float* loss_out = xq_out + (size_t)N_ROWS * E_DIM;     // [8388608]
    float* idx_out  = loss_out + 1;                        // [8388609, +16384)

    vq_prep_w<<<N_E, 256, 0, stream>>>(W, Wb, Wnorm);
    vq_conv_x<<<(N_ROWS * E_DIM / 4) / 256, 256, 0, stream>>>(X, Xb);
    vq_zero<<<1, 1, 0, stream>>>(acc);
    vq_argmin<<<N_ROWS / 32, 256, 0, stream>>>(Xb, Wb, Wnorm, idxB);
    vq_gather_loss<<<N_ROWS, 256, 0, stream>>>(X, W, idxB, xq_out, idx_out, acc);
    vq_final<<<1, 1, 0, stream>>>(acc, loss_out);
}
